// BandhaAttention_36696200577509
// MI455X (gfx1250) — matrix operand order repaired and checked by reading, not compiled, not run on hardware
//
#include <hip/hip_runtime.h>

typedef __attribute__((ext_vector_type(8)))  float          v8f;
typedef __attribute__((ext_vector_type(16))) __bf16         v16bf;
typedef __attribute__((ext_vector_type(4)))  unsigned int   v4u;
typedef unsigned short ushort_t;

union Frag {
    v16bf v;
    v4u   q[2];
    ushort_t u[16];
};

__device__ __forceinline__ unsigned bf_rne(float f) {
    unsigned u = __float_as_uint(f);
    u += 0x7FFFu + ((u >> 16) & 1u);          // round-to-nearest-even
    return u >> 16;
}
__device__ __forceinline__ ushort_t f2bf(float f) { return (ushort_t)bf_rne(f); }
// pack two floats -> {hi:bf16(b), lo:bf16(a)} in one dword
__device__ __forceinline__ unsigned pk2bf(float a, float b) {
    return bf_rne(a) | (bf_rne(b) << 16);
}
__device__ __forceinline__ float sigmoidf_(float x) { return 1.f / (1.f + __expf(-x)); }

#define WMMA_BF16(A, B, C) \
    __builtin_amdgcn_wmma_f32_16x16x32_bf16(false, (A), false, (B), (short)0, (C), false, false)

// ---------------------------------------------------------------------------
// Kernel 0: gate table  Gt[h][t] = sigmoid(bandha[h, t % cyc(h)]) * 0.125
// ---------------------------------------------------------------------------
__global__ __launch_bounds__(256) void bandha_gate_kernel(
    const float* __restrict__ bandha, float* __restrict__ Gt) {
    const int idx = blockIdx.x * 256 + threadIdx.x;      // h*2048 + t
    const int h = idx >> 11, tt = idx & 2047;
    const int cyc = 5 + (h & 3);                         // TALA cycles 5,6,7,8
    Gt[idx] = sigmoidf_(bandha[h * 8 + (tt % cyc)]) * 0.125f;
}

// ---------------------------------------------------------------------------
// Shared WMMA GEMM mainloop: 256 threads = 8 waves, block tile 128x128,
// wave tile 32x64 (waves arranged 4(M) x 2(N)), k-step 32.
// A is MxK (fp32 or bf16), B is KxN fp32 (converted to bf16 on stage).
// LDS layouts (bf16): As[row][k] row-major stride 32; Bs[col][k] stride 32.
// ---------------------------------------------------------------------------
template <bool A_IS_BF16>
__device__ __forceinline__ void gemm_tiles(const void* Aptr, int lda,
                                           const float* B, int ldb,
                                           int m0, int n0, int K,
                                           ushort_t* As, ushort_t* Bs,
                                           v8f c[2][4]) {
    const int t    = threadIdx.x;
    const int wave = t >> 5, lane = t & 31;
    const int wm   = wave >> 1, wn = wave & 1;
    const int l16  = lane & 15;
    const int hi   = (lane < 16) ? 0 : 8;    // C row offset
    const int khlf = (lane < 16) ? 0 : 16;   // B-frag K half

    #pragma unroll
    for (int mt = 0; mt < 2; ++mt)
        #pragma unroll
        for (int nt = 0; nt < 4; ++nt)
            #pragma unroll
            for (int i = 0; i < 8; ++i) c[mt][nt][i] = 0.f;

    const int rowA = t >> 1, halfA = (t & 1) * 16;   // A staging: 128 rows x 32 k
    const int kpB  = t >> 4, csB   = (t & 15) * 8;   // B staging: 16 k-pairs x 128 cols
    unsigned* Bsu = (unsigned*)Bs;                   // Bsu[col*16 + kpair]

    for (int k0 = 0; k0 < K; k0 += 32) {
        // ---- stage A tile (128x32) into LDS as packed bf16 pairs (2 x b128)
        {
            ushort_t* dst = As + rowA * 32 + halfA;
            if (A_IS_BF16) {
                const ushort_t* src = (const ushort_t*)Aptr + (size_t)(m0 + rowA) * lda + k0 + halfA;
                *(v4u*)dst       = *(const v4u*)src;
                *(v4u*)(dst + 8) = *(const v4u*)(src + 8);
                if (k0 + 32 < K) __builtin_prefetch(src + 32, 0, 1);
            } else {
                const float* src = (const float*)Aptr + (size_t)(m0 + rowA) * lda + k0 + halfA;
                v4u p0, p1;
                #pragma unroll
                for (int j = 0; j < 4; ++j) p0[j] = pk2bf(src[2 * j],     src[2 * j + 1]);
                #pragma unroll
                for (int j = 0; j < 4; ++j) p1[j] = pk2bf(src[8 + 2 * j], src[9 + 2 * j]);
                *(v4u*)dst       = p0;
                *(v4u*)(dst + 8) = p1;
                if (k0 + 32 < K) __builtin_prefetch(src + 32, 0, 1);
            }
        }
        // ---- stage B tile (32x128) transposed+packed: 8 x b32 stores
        {
            const float* src0 = B + (size_t)(k0 + 2 * kpB)     * ldb + n0 + csB;
            const float* src1 = B + (size_t)(k0 + 2 * kpB + 1) * ldb + n0 + csB;
            #pragma unroll
            for (int i = 0; i < 8; ++i)
                Bsu[(csB + i) * 16 + kpB] = pk2bf(src0[i], src1[i]);
            if (k0 + 32 < K) {
                __builtin_prefetch(src0 + 32 * ldb, 0, 1);
                __builtin_prefetch(src1 + 32 * ldb, 0, 1);
            }
        }
        __syncthreads();

        Frag a[2];
        #pragma unroll
        for (int mt = 0; mt < 2; ++mt) {
            const ushort_t* p = As + (wm * 32 + mt * 16 + l16) * 32;
            a[mt].q[0] = *(const v4u*)(p + khlf);      // K = khlf..khlf+7
            a[mt].q[1] = *(const v4u*)(p + khlf + 8);  // K = khlf+8..khlf+15
        }
        #pragma unroll
        for (int nt = 0; nt < 4; ++nt) {
            Frag b;
            const ushort_t* p = Bs + (wn * 64 + nt * 16 + l16) * 32 + khlf;
            b.q[0] = *(const v4u*)p;                    // K = khlf..khlf+7
            b.q[1] = *(const v4u*)(p + 8);              // K = khlf+8..khlf+15
            #pragma unroll
            for (int mt = 0; mt < 2; ++mt)
                c[mt][nt] = WMMA_BF16(a[mt].v, b.v, c[mt][nt]);
        }
        __syncthreads();
    }
}

// ---------------------------------------------------------------------------
// Kernel 1: qkv = x @ w_qkv ; epilogue scatters gated Q, K, V^T (bf16)
//   Qg: (B,H,T,64)  scaled by Gt[h][t] (sigmoid gate * 1/sqrt(hd))
//   Kg: (B,H,T,64)
//   Vt: (B,H,64,T)  transposed so PV B-fragments are contiguous
// ---------------------------------------------------------------------------
__global__ __launch_bounds__(256) void bandha_qkv_kernel(
    const float* __restrict__ x, const float* __restrict__ w_qkv,
    const float* __restrict__ Gt,
    ushort_t* __restrict__ Qg, ushort_t* __restrict__ Kg, ushort_t* __restrict__ Vt) {
    __shared__ ushort_t As[128 * 32];
    __shared__ ushort_t Bs[128 * 32];
    v8f c[2][4];
    const int m0 = blockIdx.y * 128, n0 = blockIdx.x * 128;
    gemm_tiles<false>(x, 1024, w_qkv, 3072, m0, n0, 1024, As, Bs, c);

    const int t = threadIdx.x, wave = t >> 5, lane = t & 31;
    const int wm = wave >> 1, wn = wave & 1, l16 = lane & 15;
    const int hi = (lane < 16) ? 0 : 8;
    #pragma unroll
    for (int mt = 0; mt < 2; ++mt) {
        #pragma unroll
        for (int i = 0; i < 8; ++i) {
            const int m  = m0 + wm * 32 + mt * 16 + hi + i;   // b*T + t
            const int bb = m >> 11, tt = m & 2047;
            #pragma unroll
            for (int nt = 0; nt < 4; ++nt) {
                const int n      = n0 + wn * 64 + nt * 16 + l16;
                const float val  = c[mt][nt][i];
                const int chunk  = n >> 10;          // 0=Q 1=K 2=V
                const int within = n & 1023;
                const int h = within >> 6, d = within & 63;
                const size_t hb = (size_t)(bb * 16 + h);
                if (chunk == 0) {
                    Qg[(hb * 2048 + tt) * 64 + d] = f2bf(val * Gt[h * 2048 + tt]);
                } else if (chunk == 1) {
                    Kg[(hb * 2048 + tt) * 64 + d] = f2bf(val);
                } else {
                    Vt[(hb * 64 + d) * 2048 + tt] = f2bf(val);
                }
            }
        }
    }
}

// ---------------------------------------------------------------------------
// Kernel 2: causal flash attention. 4 waves/block, 16 query rows per wave,
// key tiles of 64. All matmuls via v_wmma_f32_16x16x32_bf16.
// ---------------------------------------------------------------------------
__global__ __launch_bounds__(128) void bandha_attn_kernel(
    const ushort_t* __restrict__ Qg, const ushort_t* __restrict__ Kg,
    const ushort_t* __restrict__ Vt, ushort_t* __restrict__ O) {
    __shared__ ushort_t Pl[4 * 16 * 64];   // per-wave P tiles (C-layout -> A-layout relay)

    const int t = threadIdx.x, wave = t >> 5, lane = t & 31;
    const int l16  = lane & 15;
    const int hi   = (lane < 16) ? 0 : 8;
    const int khlf = (lane < 16) ? 0 : 16;
    const int bh   = blockIdx.y;                     // b*16 + h
    const int qbase = blockIdx.x * 64 + wave * 16;

    const ushort_t* Qh = Qg + (size_t)bh * 2048 * 64;
    const ushort_t* Kh = Kg + (size_t)bh * 2048 * 64;
    const ushort_t* Vh = Vt + (size_t)bh * 64 * 2048;

    // Q A-fragments for this wave's 16 rows (hd = 64 -> two k-steps)
    Frag aq[2];
    {
        const ushort_t* qp = Qh + (size_t)(qbase + l16) * 64;
        #pragma unroll
        for (int kk = 0; kk < 2; ++kk) {
            aq[kk].q[0] = *(const v4u*)(qp + kk * 32 + khlf);
            aq[kk].q[1] = *(const v4u*)(qp + kk * 32 + khlf + 8);
        }
    }

    float m_i[8], l_i[8];
    v8f acc[4];
    #pragma unroll
    for (int i = 0; i < 8; ++i) { m_i[i] = -__builtin_inff(); l_i[i] = 0.f; }
    #pragma unroll
    for (int nt = 0; nt < 4; ++nt)
        #pragma unroll
        for (int i = 0; i < 8; ++i) acc[nt][i] = 0.f;

    ushort_t* Pw = Pl + wave * 1024;
    const int ntiles = (qbase + 15) / 64 + 1;        // causal per-wave bound

    for (int ti = 0; ti < ntiles; ++ti) {
        const int kb = ti * 64;

        // ---- S = Q @ K^T (pre-scaled); K B-frags contiguous from (T,hd) layout
        v8f s[4];
        #pragma unroll
        for (int nt = 0; nt < 4; ++nt) {
            Frag bk0, bk1;
            const ushort_t* kp = Kh + (size_t)(kb + nt * 16 + l16) * 64 + khlf;
            bk0.q[0] = *(const v4u*)(kp);       bk0.q[1] = *(const v4u*)(kp + 8);
            bk1.q[0] = *(const v4u*)(kp + 32);  bk1.q[1] = *(const v4u*)(kp + 40);
            v8f z;
            #pragma unroll
            for (int i = 0; i < 8; ++i) z[i] = 0.f;
            z = WMMA_BF16(aq[0].v, bk0.v, z);
            z = WMMA_BF16(aq[1].v, bk1.v, z);
            s[nt] = z;
        }

        // ---- causal mask + online softmax (row stats across 16-lane halves)
        #pragma unroll
        for (int i = 0; i < 8; ++i) {
            const int row = qbase + hi + i;
            float mx = -__builtin_inff();
            #pragma unroll
            for (int nt = 0; nt < 4; ++nt) {
                const int col = kb + nt * 16 + l16;
                const float v = (col <= row) ? s[nt][i] : -__builtin_inff();
                s[nt][i] = v;
                mx = fmaxf(mx, v);
            }
            #pragma unroll
            for (int d = 1; d < 16; d <<= 1) mx = fmaxf(mx, __shfl_xor(mx, d, 32));
            const float mnew  = fmaxf(m_i[i], mx);
            const float alpha = __expf(m_i[i] - mnew);
            float rs = 0.f;
            #pragma unroll
            for (int nt = 0; nt < 4; ++nt) {
                const float p = __expf(s[nt][i] - mnew);
                s[nt][i] = p;
                rs += p;
            }
            #pragma unroll
            for (int d = 1; d < 16; d <<= 1) rs += __shfl_xor(rs, d, 32);
            l_i[i] = l_i[i] * alpha + rs;
            m_i[i] = mnew;
            #pragma unroll
            for (int nt = 0; nt < 4; ++nt) acc[nt][i] *= alpha;
        }

        // ---- relay P through wave-private LDS: C-layout -> A-layout
        #pragma unroll
        for (int nt = 0; nt < 4; ++nt)
            #pragma unroll
            for (int i = 0; i < 8; ++i)
                Pw[(hi + i) * 64 + nt * 16 + l16] = f2bf(s[nt][i]);

        Frag ap[2];
        {
            const ushort_t* pp = Pw + l16 * 64;
            #pragma unroll
            for (int kk = 0; kk < 2; ++kk) {
                ap[kk].q[0] = *(const v4u*)(pp + kk * 32 + khlf);
                ap[kk].q[1] = *(const v4u*)(pp + kk * 32 + khlf + 8);
            }
        }

        // ---- acc += P @ V ; V^T layout makes B-frags contiguous
        #pragma unroll
        for (int nt = 0; nt < 4; ++nt) {
            Frag bv0, bv1;
            const ushort_t* vp = Vh + (size_t)(nt * 16 + l16) * 2048 + kb + khlf;
            bv0.q[0] = *(const v4u*)(vp);       bv0.q[1] = *(const v4u*)(vp + 8);
            bv1.q[0] = *(const v4u*)(vp + 32);  bv1.q[1] = *(const v4u*)(vp + 40);
            acc[nt] = WMMA_BF16(ap[0].v, bv0.v, acc[nt]);
            acc[nt] = WMMA_BF16(ap[1].v, bv1.v, acc[nt]);
        }
    }

    // ---- normalize and write O as (B, T, H, hd) bf16 (input of out-proj GEMM)
    const int bb = bh >> 4, h = bh & 15;
    #pragma unroll
    for (int i = 0; i < 8; ++i) {
        const float inv = 1.f / l_i[i];
        const int row = qbase + hi + i;
        #pragma unroll
        for (int nt = 0; nt < 4; ++nt)
            O[((size_t)(bb * 2048 + row) * 16 + h) * 64 + nt * 16 + l16] =
                f2bf(acc[nt][i] * inv);
    }
}

// ---------------------------------------------------------------------------
// Kernel 3: out = O @ w_out (fp32 result)
// ---------------------------------------------------------------------------
__global__ __launch_bounds__(256) void bandha_out_kernel(
    const ushort_t* __restrict__ O, const float* __restrict__ w_out,
    float* __restrict__ out) {
    __shared__ ushort_t As[128 * 32];
    __shared__ ushort_t Bs[128 * 32];
    v8f c[2][4];
    const int m0 = blockIdx.y * 128, n0 = blockIdx.x * 128;
    gemm_tiles<true>(O, 1024, w_out, 1024, m0, n0, 1024, As, Bs, c);

    const int t = threadIdx.x, wave = t >> 5, lane = t & 31;
    const int wm = wave >> 1, wn = wave & 1, l16 = lane & 15;
    const int hi = (lane < 16) ? 0 : 8;
    #pragma unroll
    for (int mt = 0; mt < 2; ++mt)
        #pragma unroll
        for (int i = 0; i < 8; ++i) {
            const int m = m0 + wm * 32 + mt * 16 + hi + i;
            #pragma unroll
            for (int nt = 0; nt < 4; ++nt) {
                const int n = n0 + wn * 64 + nt * 16 + l16;
                out[(size_t)m * 1024 + n] = c[mt][nt][i];
            }
        }
}

// ---------------------------------------------------------------------------
extern "C" void kernel_launch(void* const* d_in, const int* in_sizes, int n_in,
                              void* d_out, int out_size, void* d_ws, size_t ws_size,
                              hipStream_t stream) {
    const float* x      = (const float*)d_in[0];   // (2,2048,1024)
    const float* w_qkv  = (const float*)d_in[1];   // (1024,3072)
    const float* w_out  = (const float*)d_in[2];   // (1024,1024)
    const float* bandha = (const float*)d_in[3];   // (16,8)
    float* out = (float*)d_out;

    float* Gt = (float*)d_ws;                            // 16*2048 fp32 gate table
    const size_t HEADBUF = (size_t)2 * 16 * 2048 * 64;   // 4,194,304 bf16 elements
    ushort_t* Qg = (ushort_t*)(Gt + 16 * 2048);
    ushort_t* Kg = Qg + HEADBUF;
    ushort_t* Vt = Kg + HEADBUF;
    ushort_t* O  = Vt + HEADBUF;                         // ~32.5 MB total

    // gate table: sigmoid(bandha[h, t % cyc]) * (1/sqrt(64))
    bandha_gate_kernel<<<dim3(128), 256, 0, stream>>>(bandha, Gt);
    // qkv projection + gate/scale/scatter:  M=4096, N=3072, K=1024
    bandha_qkv_kernel<<<dim3(24, 32), 256, 0, stream>>>(x, w_qkv, Gt, Qg, Kg, Vt);
    // causal flash attention: grid (T/64, B*H)
    bandha_attn_kernel<<<dim3(32, 32), 128, 0, stream>>>(Qg, Kg, Vt, O);
    // output projection: M=4096, N=1024, K=1024
    bandha_out_kernel<<<dim3(8, 32), 256, 0, stream>>>(O, w_out, out);
}
